// VQEmbedding_32323923870348
// MI455X (gfx1250) — matrix operand order repaired and checked by reading, not compiled, not run on hardware
//
#include <hip/hip_runtime.h>

// ---------------- types ----------------
typedef _Float16 v16h __attribute__((ext_vector_type(16)));
typedef _Float16 v8h  __attribute__((ext_vector_type(8)));
typedef _Float16 v4h  __attribute__((ext_vector_type(4)));
typedef float    v8f  __attribute__((ext_vector_type(8)));

#define NEMB    1024
#define NEMBDIM 256
#define NTOK    65536
#define MBLK    64      // token rows per workgroup
#define NBLK    32      // codes per inner tile (2 waves x 16)

// ---------------- prep: emb f32 -> f16, e-norms ----------------
__global__ __launch_bounds__(256)
void vq_prep(const float* __restrict__ emb,
             _Float16* __restrict__ emb_h,
             float* __restrict__ enorm) {
    __shared__ float red[256];
    const int row = blockIdx.x;
    const int d   = threadIdx.x;
    float v = emb[row * NEMBDIM + d];
    emb_h[row * NEMBDIM + d] = (_Float16)v;
    red[d] = v * v;
    __syncthreads();
    for (int s = 128; s > 0; s >>= 1) {
        if (d < s) red[d] += red[d + s];
        __syncthreads();
    }
    if (d == 0) enorm[row] = red[0];
}

// ---------------- main: scores via WMMA, argmin, gather, loss partial ----------------
__global__ __launch_bounds__(256)
void vq_main(const float* __restrict__ x,
             const float* __restrict__ emb,
             const _Float16* __restrict__ emb_h,
             const float* __restrict__ enorm,
             float* __restrict__ out,
             float* __restrict__ partial) {
    __shared__ __align__(32) _Float16 At[MBLK][NEMBDIM];      // 32 KB x-tile (f16)
    __shared__ __align__(32) _Float16 Bt[2][NBLK][NEMBDIM];   // 2 x 16 KB code tiles
    __shared__ float s_minv[MBLK][2];
    __shared__ int   s_mini[MBLK][2];
    __shared__ int   s_idx[MBLK];
    __shared__ float s_red[256];

    const int tid     = threadIdx.x;
    const int rowbase = blockIdx.x * MBLK;

    // ---- stage A tile: 64 x 256 f32 -> f16 in LDS ----
    {
        const float4* xv = (const float4*)(x + (size_t)rowbase * NEMBDIM);
        #pragma unroll
        for (int j = 0; j < 16; ++j) {
            int f4 = tid + j * 256;           // 0..4095 float4 chunks
            float4 v = xv[f4];
            v4h h; h.x = (_Float16)v.x; h.y = (_Float16)v.y;
                   h.z = (_Float16)v.z; h.w = (_Float16)v.w;
            int r = f4 >> 6, c4 = f4 & 63;    // 64 float4 per row
            *(v4h*)&At[r][c4 * 4] = h;
        }
    }

    const int wave = tid >> 5;      // 0..7
    const int lane = tid & 31;
    const int l15  = lane & 15;
    const int hi   = lane >> 4;     // lane half (selects K sub-chunk)
    const int mrow = wave >> 1;     // 0..3 : 16-row tile
    const int ncol = wave & 1;      // 0..1 : 16-code tile

    __syncthreads();                // At fully staged

    // ---- load ALL A fragments into registers once (loop-invariant) ----
    v16h afrag[8];
    #pragma unroll
    for (int kc = 0; kc < 8; ++kc) {
        // A 16x32 layout (matches B): lane half hi -> K = kc*32 + hi*16 + {0..15} contiguous
        v8h alo = *(const v8h*)&At[mrow * 16 + l15][kc * 32 + hi * 16];
        v8h ahi = *(const v8h*)&At[mrow * 16 + l15][kc * 32 + hi * 16 + 8];
        afrag[kc] = __builtin_shufflevector(alo, ahi,
                    0,1,2,3,4,5,6,7,8,9,10,11,12,13,14,15);
    }

    // Per-LANE running argmin: each lane sees one code column per tile; its
    // column indices strictly increase with t, so strict < keeps the first
    // (lowest) index on ties — matching jnp.argmin.
    float runv[8];
    int   runi[8];
    #pragma unroll
    for (int r = 0; r < 8; ++r) { runv[r] = 3.4e38f; runi[r] = 0; }

    // ---- prefetch first B tile into registers ----
    const uint4* bsrc = (const uint4*)emb_h;   // 512 uint4 (16 codes) per tile step
    uint4 g0 = bsrc[tid];
    uint4 g1 = bsrc[tid + 256];

    for (int t = 0; t < 32; ++t) {
        // store current tile into LDS buffer t&1
        {
            uint4* dst = (uint4*)&Bt[t & 1][0][0];
            dst[tid]       = g0;
            dst[tid + 256] = g1;
        }
        // prefetch next tile (clamped; overlaps with this tile's compute)
        {
            int tn = (t < 31) ? t + 1 : 31;
            const uint4* src = bsrc + (size_t)tn * 512;
            g0 = src[tid];
            g1 = src[tid + 256];
        }
        __syncthreads();   // Bt[t&1] visible; also WAR fence for buffer reuse (2 iters apart)

        const _Float16 (*B)[NEMBDIM] = Bt[t & 1];

        // ---- 16x16 score tile, K=256 in 8 WMMA steps (A in regs, B from LDS) ----
        v8f acc = {};
        #pragma unroll
        for (int kc = 0; kc < 8; ++kc) {
            // B frag: lane = code column, K = kc*32 + hi*16 + {0..15} contiguous
            v16h b = *(const v16h*)&B[ncol * 16 + l15][kc * 32 + hi * 16];
            acc = __builtin_amdgcn_wmma_f32_16x16x32_f16(
                      false, afrag[kc], false, b, (short)0, acc, false, false);
        }

        // ---- dist = ||e||^2 - score ; branch-free per-lane running min ----
        const int ng = t * 32 + ncol * 16 + l15;     // global code index for this lane
        const float en = enorm[ng];
        #pragma unroll
        for (int r = 0; r < 8; ++r) {
            float v = en - acc[r];
            bool lt = v < runv[r];
            runv[r] = lt ? v  : runv[r];
            runi[r] = lt ? ng : runi[r];
        }
    }

    // ---- ONE cross-lane butterfly after the loop (16 lanes per row) ----
    #pragma unroll
    for (int r = 0; r < 8; ++r) {
        float v = runv[r];
        int   i = runi[r];
        #pragma unroll
        for (int m = 1; m < 16; m <<= 1) {
            float ov = __shfl_xor(v, m, 32);
            int   oi = __shfl_xor(i, m, 32);
            int take = (int)(ov < v) | ((int)(ov == v) & (int)(oi < i));
            v = take ? ov : v;
            i = take ? oi : i;
        }
        runv[r] = v;
        runi[r] = i;
    }

    // ---- combine the two code-column waves per row ----
    if (l15 == 0) {
        #pragma unroll
        for (int r = 0; r < 8; ++r) {
            int m = mrow * 16 + hi * 8 + r;
            s_minv[m][ncol] = runv[r];
            s_mini[m][ncol] = runi[r];
        }
    }
    __syncthreads();
    if (tid < MBLK) {
        float v0 = s_minv[tid][0], v1 = s_minv[tid][1];
        int   i0 = s_mini[tid][0], i1 = s_mini[tid][1];
        int take = (int)(v1 < v0) | ((int)(v1 == v0) & (int)(i1 < i0));
        s_idx[tid] = take ? i1 : i0;
    }
    __syncthreads();

    // ---- gather f32 codebook rows, emit out = x + (e - x), accumulate loss ----
    float sum = 0.f;
    {
        const int r = tid >> 2, q = tid & 3;      // 4 threads per row, 64 cols each
        const int row = rowbase + r;
        const int idx = s_idx[r];
        const float4* ev = (const float4*)(emb + (size_t)idx * NEMBDIM);
        const float4* xv = (const float4*)(x   + (size_t)row * NEMBDIM);
        float4* ov = (float4*)(out + (size_t)row * NEMBDIM);
        #pragma unroll
        for (int j = 0; j < 16; ++j) {
            int c4 = q * 16 + j;
            float4 e  = ev[c4];
            float4 xx = xv[c4];
            float dx = e.x - xx.x, dy = e.y - xx.y, dz = e.z - xx.z, dw = e.w - xx.w;
            float4 o;
            o.x = xx.x + dx; o.y = xx.y + dy; o.z = xx.z + dz; o.w = xx.w + dw;
            ov[c4] = o;
            sum += dx * dx + dy * dy + dz * dz + dw * dw;
        }
    }
    s_red[tid] = sum;
    __syncthreads();
    for (int s = 128; s > 0; s >>= 1) {
        if (tid < s) s_red[tid] += s_red[tid + s];
        __syncthreads();
    }
    if (tid == 0) partial[blockIdx.x] = s_red[0];
}

// ---------------- final loss reduction (deterministic, single WG) ----------------
__global__ __launch_bounds__(256)
void vq_loss(const float* __restrict__ partial, float* __restrict__ out_loss) {
    __shared__ float red[256];
    const int tid = threadIdx.x;
    float s = 0.f;
    #pragma unroll
    for (int j = 0; j < 4; ++j) s += partial[tid + j * 256];
    red[tid] = s;
    __syncthreads();
    for (int st = 128; st > 0; st >>= 1) {
        if (tid < st) red[tid] += red[tid + st];
        __syncthreads();
    }
    if (tid == 0)
        out_loss[0] = red[0] * (1.25f / ((float)NTOK * (float)NEMBDIM));
}

// ---------------- host launcher ----------------
extern "C" void kernel_launch(void* const* d_in, const int* in_sizes, int n_in,
                              void* d_out, int out_size, void* d_ws, size_t ws_size,
                              hipStream_t stream) {
    (void)in_sizes; (void)n_in; (void)out_size; (void)ws_size;
    const float* x   = (const float*)d_in[0];   // [NTOK, NEMBDIM] f32
    const float* emb = (const float*)d_in[1];   // [NEMB, NEMBDIM] f32
    float* out = (float*)d_out;                 // [NTOK*NEMBDIM] + 1 loss scalar

    // workspace layout: emb_f16 (512KB) | enorm (4KB) | partials (4KB)
    _Float16* emb_h = (_Float16*)d_ws;
    float* enorm    = (float*)((char*)d_ws + (size_t)NEMB * NEMBDIM * sizeof(_Float16));
    float* partial  = enorm + NEMB;

    vq_prep<<<NEMB, 256, 0, stream>>>(emb, emb_h, enorm);
    vq_main<<<NTOK / MBLK, 256, 0, stream>>>(x, emb, emb_h, enorm, out, partial);
    vq_loss<<<1, 256, 0, stream>>>(partial, out + (size_t)NTOK * NEMBDIM);
}
